// BronxLayer_3805341024700
// MI455X (gfx1250) — matrix operand order repaired and checked by reading, not compiled, not run on hardware
//
#include <hip/hip_runtime.h>

#define NN   3072
#define HH   256
#define NHD  4
#define HD   64
#define LN_EPS 1e-5f

typedef __attribute__((ext_vector_type(16))) __bf16 v16bf;
typedef __attribute__((ext_vector_type(8)))  float  v8f;
typedef __attribute__((ext_vector_type(4)))  unsigned su4;
typedef __attribute__((ext_vector_type(8)))  unsigned su8;
typedef unsigned short bfr;

union FragU { v16bf v; uint4 q[2]; };

__device__ __forceinline__ bfr f2bf(float f) {
  unsigned u = __float_as_uint(f);
  u += 0x7FFFu + ((u >> 16) & 1u);          // round-to-nearest-even
  return (bfr)(u >> 16);
}

__device__ __forceinline__ v16bf load_frag(const bfr* p0, const bfr* p1) {
  FragU f;
  f.q[0] = *(const uint4*)p0;
  f.q[1] = *(const uint4*)p1;
  return f.v;
}

__device__ __forceinline__ v8f vzero() {
  v8f z;
#pragma unroll
  for (int i = 0; i < 8; ++i) z[i] = 0.f;
  return z;
}

__device__ __forceinline__ v8f bmma(v16bf a, v16bf b, v8f c) {
  return __builtin_amdgcn_wmma_f32_16x16x32_bf16(false, a, false, b, (short)0, c,
                                                 false, false);
}

// TDM: linear 32KB global -> LDS copy (1 descriptor, data_size=8B, 1-D tile).
// D# layout per CDNA5 ISA. Tracked by TENSORcnt.
__device__ __forceinline__ void tdm_copy_32k(const bfr* gsrc, bfr* lds_dst) {
  unsigned long long ga = (unsigned long long)(size_t)gsrc;
  su4 g0;
  g0[0] = 1u;                                   // count=1, user mode
  g0[1] = (unsigned)(size_t)lds_dst;            // LDS byte address (low 32 bits)
  g0[2] = (unsigned)ga;                         // global_addr[31:0]
  g0[3] = (unsigned)(ga >> 32) | (2u << 30);    // global_addr[56:32] | type=2
  su8 g1;
  const unsigned n8 = 4096u;                    // 32KB in 8-byte units
  g1[0] = 3u << 16;                             // data_size = 8B
  g1[1] = (n8 & 0xFFFFu) << 16;                 // tensor_dim0[15:0]
  g1[2] = (n8 >> 16) | (1u << 16);              // tensor_dim0[31:16] | tensor_dim1=1
  g1[3] = (n8 & 0xFFFFu) << 16;                 // tile_dim0 = 4096
  g1[4] = 1u;                                   // tile_dim1 = 1, tile_dim2 = 0
  g1[5] = n8;                                   // tensor_dim0_stride[31:0]
  g1[6] = 0u;
  g1[7] = 0u;
  asm volatile("tensor_load_to_lds %0, %1" :: "s"(g0), "s"(g1) : "memory");
}

// ---------------------------------------------------------------- LayerNorm
__global__ void k_layernorm(const float* __restrict__ h, const float* __restrict__ g,
                            const float* __restrict__ bta,
                            bfr* __restrict__ hn_bf, bfr* __restrict__ hnT) {
  int wave = threadIdx.x >> 5, lane = threadIdx.x & 31;
  int row = blockIdx.x * 8 + wave;
  const float4* hr = (const float4*)(h + (size_t)row * HH);
  float4 a = hr[lane], b = hr[lane + 32];
  float s  = a.x + a.y + a.z + a.w + b.x + b.y + b.z + b.w;
  float s2 = a.x*a.x + a.y*a.y + a.z*a.z + a.w*a.w
           + b.x*b.x + b.y*b.y + b.z*b.z + b.w*b.w;
#pragma unroll
  for (int m = 1; m < 32; m <<= 1) { s += __shfl_xor(s, m, 32); s2 += __shfl_xor(s2, m, 32); }
  float mean = s * (1.f / HH);
  float rstd = rsqrtf(s2 * (1.f / HH) - mean * mean + LN_EPS);
  float va[8] = {a.x, a.y, a.z, a.w, b.x, b.y, b.z, b.w};
#pragma unroll
  for (int i = 0; i < 8; ++i) {
    int col = (i < 4) ? (lane * 4 + i) : (128 + lane * 4 + (i - 4));
    float vv = (va[i] - mean) * rstd * g[col] + bta[col];
    bfr bb = f2bf(vv);
    hn_bf[(size_t)row * HH + col] = bb;
    hnT[(size_t)col * NN + row]  = bb;      // K-contiguous copy for WMMA B operand
  }
}

// ------------------------------------------------- weights -> transposed bf16
__global__ void k_wconv(const float* __restrict__ Wk, const float* __restrict__ Wmu,
                        const float* __restrict__ Wls, const float* __restrict__ Wv,
                        bfr* __restrict__ WT, bfr* __restrict__ WvT) {
  int i = blockIdx.x * 256 + threadIdx.x;          // 262144 threads
  int in = i >> 8, o = i & 255;
  WvT[(size_t)o * 1024 + in] = f2bf(Wv[(size_t)in * 256 + o]);
  if (i < 256 * 256) {
    WT[(size_t)o * 256 + in]          = f2bf(Wk[(size_t)in * 256 + o]);
    WT[65536 + (size_t)o * 256 + in]  = f2bf(Wmu[(size_t)in * 256 + o]);
    WT[131072 + (size_t)o * 256 + in] = f2bf(Wls[(size_t)in * 256 + o]);
  }
}

// -------------------------------------- k/mu/ls projections into head layout
__global__ void __launch_bounds__(128) k_proj(const bfr* __restrict__ hn_bf,
                                              const bfr* __restrict__ WT,
                                              const float* __restrict__ bk,
                                              const float* __restrict__ bmu,
                                              const float* __restrict__ bls,
                                              bfr* __restrict__ khead,
                                              bfr* __restrict__ muhead,
                                              bfr* __restrict__ lshead) {
  int p = blockIdx.y;
  const bfr* Wp = WT + (size_t)p * 65536;
  const float* bias = (p == 0) ? bk : ((p == 1) ? bmu : bls);
  bfr* dst = (p == 0) ? khead : ((p == 1) ? muhead : lshead);
  float scl = (p == 0) ? 0.125f : 1.0f;    // fold HD^-0.5 into k once
  int x0 = blockIdx.x * 16;
  int w = threadIdx.x >> 5, lo = threadIdx.x & 15, hi = (threadIdx.x >> 4) & 1;
  v8f acc[4];
#pragma unroll
  for (int ct = 0; ct < 4; ++ct) acc[ct] = vzero();
  for (int k0 = 0; k0 < HH; k0 += 32) {
    const bfr* ar = hn_bf + (size_t)(x0 + lo) * HH + k0;
    v16bf a = load_frag(ar + 16 * hi, ar + 16 * hi + 8);
#pragma unroll
    for (int ct = 0; ct < 4; ++ct) {
      const bfr* br = Wp + (size_t)(w * 64 + ct * 16 + lo) * HH + k0 + 16 * hi;
      acc[ct] = bmma(a, load_frag(br, br + 8), acc[ct]);
    }
  }
#pragma unroll
  for (int ct = 0; ct < 4; ++ct) {
    int col = w * 64 + ct * 16 + lo;
    float bv = bias[col];
#pragma unroll
    for (int r = 0; r < 8; ++r) {
      int row = x0 + r + 8 * hi;
      float val = (acc[ct][r] + bv) * scl;
      // reshape(N, HD, B): column col -> (d = col>>2, head = col&3)
      dst[((size_t)(col & 3) * NN + row) * HD + (col >> 2)] = f2bf(val);
    }
  }
}

// ------------------------- fused scores + KL + sample + normalize-aggregate
__global__ void __launch_bounds__(256) k_attn(const bfr* __restrict__ khead,
                                              const bfr* __restrict__ muhead,
                                              const bfr* __restrict__ lshead,
                                              const bfr* __restrict__ hnT,
                                              const float* __restrict__ diffusion,
                                              const float* __restrict__ eps,
                                              bfr* __restrict__ u_bf,
                                              float* __restrict__ klp) {
  // Double-buffered TDM staging of mu/ls 256-row super-block tiles (32KB each)
  __shared__ __align__(16) bfr lds_mu[2][256][HD];   // 64KB
  __shared__ __align__(16) bfr lds_ls[2][256][HD];   // 64KB
  __shared__ __align__(16) bfr lds_a[8][16][32];     // 8KB  (C-layout -> A-layout)
  __shared__ float lds_den[16];
  __shared__ float lds_kl;
  int tid = threadIdx.x;
  if (tid < 16) lds_den[tid] = 0.f;
  if (tid == 0) lds_kl = 0.f;
  int x0 = blockIdx.x * 16, bh = blockIdx.y;
  int w = tid >> 5, lo = tid & 15, hi = (tid >> 4) & 1;
  const bfr* kb = khead  + (size_t)bh * NN * HD;
  const bfr* mb = muhead + (size_t)bh * NN * HD;
  const bfr* lb = lshead + (size_t)bh * NN * HD;

  // Prologue: wave 0 DMAs super-block 0 into buffer 0.
  if (w == 0) {
    tdm_copy_32k(mb, &lds_mu[0][0][0]);
    tdm_copy_32k(lb, &lds_ls[0][0][0]);
    __builtin_amdgcn_s_wait_tensorcnt(0);
  }
  __syncthreads();

  // A operand (K rows for this x-tile) is invariant over the whole y loop
  const bfr* kr = kb + (size_t)(x0 + lo) * HD;
  v16bf ak0 = load_frag(kr + 16 * hi, kr + 16 * hi + 8);       // d = 0..31
  v16bf ak1 = load_frag(kr + 32 + 16 * hi, kr + 40 + 16 * hi);  // d = 32..63
  v8f oacc[2]; oacc[0] = vzero(); oacc[1] = vzero();
  float dsum[8];
#pragma unroll
  for (int r = 0; r < 8; ++r) dsum[r] = 0.f;
  float klacc = 0.f;

  int cur = 0;
  for (int y0 = 0; y0 < NN; y0 += 256) {
    bool more = (y0 + 256) < NN;
    if (w == 0 && more) {   // overlap next super-block DMA with compute
      tdm_copy_32k(mb + (size_t)(y0 + 256) * HD, &lds_mu[cur ^ 1][0][0]);
      tdm_copy_32k(lb + (size_t)(y0 + 256) * HD, &lds_ls[cur ^ 1][0][0]);
    }
    if (more) {             // warm L2 for next eps tile
      size_t pe = ((size_t)(x0 + 8 * hi) * NN + (y0 + 256 + w * 32 + lo));
      __builtin_prefetch(eps + pe * 4 + bh, 0, 1);
      __builtin_prefetch(diffusion + pe, 0, 1);
    }
    v8f mus[2], lss[2];
#pragma unroll
    for (int yt = 0; yt < 2; ++yt) { mus[yt] = vzero(); lss[yt] = vzero(); }
#pragma unroll
    for (int yt = 0; yt < 2; ++yt) {
      const bfr* mr = &lds_mu[cur][w * 32 + yt * 16 + lo][16 * hi];
      const bfr* lr = &lds_ls[cur][w * 32 + yt * 16 + lo][16 * hi];
      mus[yt] = bmma(ak0, load_frag(mr, mr + 8), mus[yt]);
      mus[yt] = bmma(ak1, load_frag(mr + 32, mr + 40), mus[yt]);
      lss[yt] = bmma(ak0, load_frag(lr, lr + 8), lss[yt]);
      lss[yt] = bmma(ak1, load_frag(lr + 32, lr + 40), lss[yt]);
    }
#pragma unroll
    for (int yt = 0; yt < 2; ++yt) {
      int y = y0 + w * 32 + yt * 16 + lo;
#pragma unroll
      for (int r = 0; r < 8; ++r) {
        int xr = x0 + r + 8 * hi;
        size_t e2 = (size_t)xr * NN + y;
        float diff = diffusion[e2];
        float ep   = eps[e2 * 4 + bh];
        float mu   = mus[yt][r];
        float lsv  = lss[yt][r];
        float sg = fmaxf(lsv, 0.f) + log1pf(expf(-fabsf(lsv)));   // softplus
        float sgn = (diff > 0.f ? 1.f : 0.f) - (diff < 0.f ? 1.f : 0.f);
        klacc += (0.5f * (sg * sg + mu * mu) - logf(sg) - 0.5f) * sgn;
        float av = expf(fmaf(sg, ep, mu)) * diff;                 // a >= 0
        dsum[r] += av;
        lds_a[w][r + 8 * hi][yt * 16 + lo] = f2bf(av);
      }
    }
    __syncthreads();
#pragma unroll
    for (int sb = 0; sb < 8; ++sb) {
      const bfr* pa = &lds_a[sb][lo][0];
      v16bf af = load_frag(pa + 16 * hi, pa + 16 * hi + 8);
#pragma unroll
      for (int ct = 0; ct < 2; ++ct) {
        const bfr* pb = hnT + (size_t)(w * 32 + ct * 16 + lo) * NN
                            + y0 + sb * 32 + 16 * hi;
        oacc[ct] = bmma(af, load_frag(pb, pb + 8), oacc[ct]);
      }
    }
    if (w == 0 && more) __builtin_amdgcn_s_wait_tensorcnt(0);  // next tiles landed
    __syncthreads();
    cur ^= 1;
  }
  // denominator: reduce over the 16 y-columns held across lanes
#pragma unroll
  for (int m = 1; m < 16; m <<= 1) {
#pragma unroll
    for (int r = 0; r < 8; ++r) dsum[r] += __shfl_xor(dsum[r], m, 32);
  }
  if (lo == 0) {
#pragma unroll
    for (int r = 0; r < 8; ++r) atomicAdd(&lds_den[r + 8 * hi], dsum[r]);
  }
#pragma unroll
  for (int m = 1; m < 32; m <<= 1) klacc += __shfl_xor(klacc, m, 32);
  if ((tid & 31) == 0) atomicAdd(&lds_kl, klacc);
  __syncthreads();
#pragma unroll
  for (int ct = 0; ct < 2; ++ct) {
    int col = w * 32 + ct * 16 + lo;
#pragma unroll
    for (int r = 0; r < 8; ++r) {
      int row = x0 + r + 8 * hi;
      float inv = 1.0f / fmaxf(lds_den[r + 8 * hi], 1e-12f);
      u_bf[(size_t)row * (NHD * HH) + (size_t)bh * HH + col] = f2bf(oacc[ct][r] * inv);
    }
  }
  if (tid == 0) atomicAdd(klp, lds_kl * (1.0f / ((float)NN * (float)NN)));
}

// ------------------------------------------------ fc_v + ELU + residual
__global__ void __launch_bounds__(128) k_fc(const bfr* __restrict__ u_bf,
                                            const bfr* __restrict__ WvT,
                                            const float* __restrict__ bv,
                                            const float* __restrict__ h0,
                                            float* __restrict__ out) {
  int x0 = blockIdx.x * 16;
  int w = threadIdx.x >> 5, lo = threadIdx.x & 15, hi = (threadIdx.x >> 4) & 1;
  v8f acc[4];
#pragma unroll
  for (int ct = 0; ct < 4; ++ct) acc[ct] = vzero();
  for (int k0 = 0; k0 < 1024; k0 += 32) {
    const bfr* ar = u_bf + (size_t)(x0 + lo) * 1024 + k0;
    v16bf a = load_frag(ar + 16 * hi, ar + 16 * hi + 8);
#pragma unroll
    for (int ct = 0; ct < 4; ++ct) {
      const bfr* br = WvT + (size_t)(w * 64 + ct * 16 + lo) * 1024 + k0 + 16 * hi;
      acc[ct] = bmma(a, load_frag(br, br + 8), acc[ct]);
    }
  }
#pragma unroll
  for (int ct = 0; ct < 4; ++ct) {
    int col = w * 64 + ct * 16 + lo;
    float bb = bv[col];
#pragma unroll
    for (int r = 0; r < 8; ++r) {
      int row = x0 + r + 8 * hi;
      float v = acc[ct][r] + bb;
      float e = v > 0.f ? v : (expf(v) - 1.f);   // ELU(alpha=1)
      out[(size_t)row * HH + col] = e + h0[(size_t)row * HH + col];
    }
  }
}

extern "C" void kernel_launch(void* const* d_in, const int* in_sizes, int n_in,
                              void* d_out, int out_size, void* d_ws, size_t ws_size,
                              hipStream_t stream) {
  const float* h     = (const float*)d_in[0];
  const float* gamma = (const float*)d_in[1];
  const float* beta  = (const float*)d_in[2];
  const float* W_k   = (const float*)d_in[3];
  const float* b_k   = (const float*)d_in[4];
  const float* W_mu  = (const float*)d_in[5];
  const float* b_mu  = (const float*)d_in[6];
  const float* W_ls  = (const float*)d_in[7];
  const float* b_ls  = (const float*)d_in[8];
  const float* W_v   = (const float*)d_in[9];
  const float* b_v   = (const float*)d_in[10];
  const float* diffusion = (const float*)d_in[11];
  const float* eps   = (const float*)d_in[12];

  char* ws = (char*)d_ws;
  size_t off = 0;
  auto alloc = [&](size_t bytes) {
    char* p = ws + off;
    off += (bytes + 255) & ~(size_t)255;
    return p;
  };
  bfr* hn_bf  = (bfr*)alloc((size_t)NN * HH * 2);
  bfr* hnT_bf = (bfr*)alloc((size_t)HH * NN * 2);
  bfr* WT     = (bfr*)alloc((size_t)3 * HH * HH * 2);
  bfr* WvT    = (bfr*)alloc((size_t)HH * 1024 * 2);
  bfr* khead  = (bfr*)alloc((size_t)NHD * NN * HD * 2);
  bfr* muhead = (bfr*)alloc((size_t)NHD * NN * HD * 2);
  bfr* lshead = (bfr*)alloc((size_t)NHD * NN * HD * 2);
  bfr* u_bf   = (bfr*)alloc((size_t)NN * 1024 * 2);
  (void)ws_size; (void)in_sizes; (void)n_in; (void)out_size;

  float* outp = (float*)d_out;
  float* klp  = outp + (size_t)NN * HH;
  hipMemsetAsync(klp, 0, sizeof(float), stream);

  k_layernorm<<<NN / 8, 256, 0, stream>>>(h, gamma, beta, hn_bf, hnT_bf);
  k_wconv<<<1024, 256, 0, stream>>>(W_k, W_mu, W_ls, W_v, WT, WvT);
  k_proj<<<dim3(NN / 16, 3), 128, 0, stream>>>(hn_bf, WT, b_k, b_mu, b_ls,
                                               khead, muhead, lshead);
  k_attn<<<dim3(NN / 16, NHD), 256, 0, stream>>>(khead, muhead, lshead, hnT_bf,
                                                 diffusion, eps, u_bf, klp);
  k_fc<<<NN / 16, 128, 0, stream>>>(u_bf, WvT, b_v, h, outp);
}
